// Neighbor_Attention_block_87866440942061
// MI455X (gfx1250) — matrix operand order repaired and checked by reading, not compiled, not run on hardware
//
#include <hip/hip_runtime.h>
#include <hip/hip_bf16.h>

// ---------------------------------------------------------------------------
// CDNA5 / gfx1250 implementation of the 2-layer graph-transformer block.
// All matmuls (QKV, attention QK^T / PV / adjV, Wo, FFN) run on
// v_wmma_f32_16x16x32_bf16 with f32 accumulation. Attention is flash-style
// (no NxN materialization) with a second un-normalized accumulator for the
// "+ adj" term. Everything stays resident in the 192MB L2.
// GEMM mainloop is software-pipelined (ping-pong LDS, 1 barrier/K-step).
// ---------------------------------------------------------------------------

typedef __bf16 bf16;
typedef __attribute__((ext_vector_type(16))) __bf16 v16bf;
typedef __attribute__((ext_vector_type(8)))  __bf16 bf16x8;
typedef __attribute__((ext_vector_type(4)))  __bf16 bf16x4;
typedef __attribute__((ext_vector_type(8)))  float  v8f;

#define DI __device__ __forceinline__

static constexpr int NB   = 8;       // batch
static constexpr int NN   = 2048;    // nodes
static constexpr int FOD  = 128;     // per-head dim
static constexpr int NH   = 4;       // heads
static constexpr int ROWS = NB * NN; // 16384

union V16 { v16bf v; bf16x8 h[2]; };

DI float rsum32(float x) {
#pragma unroll
  for (int m = 16; m >= 1; m >>= 1) x += __shfl_xor(x, m, 32);
  return x;
}
DI float rmax16(float x) {
#pragma unroll
  for (int m = 8; m >= 1; m >>= 1) x = fmaxf(x, __shfl_xor(x, m, 32));
  return x;
}
DI float rsum16(float x) {
#pragma unroll
  for (int m = 8; m >= 1; m >>= 1) x += __shfl_xor(x, m, 32);
  return x;
}

// ---------------------------------------------------------------------------
// adj build: zero + scatter (duplicates summed, matching .at[].add semantics)
// ---------------------------------------------------------------------------
__global__ void k_zero_f32(float* __restrict__ p, long n) {
  long i = (long)blockIdx.x * blockDim.x + threadIdx.x;
  long stride = (long)gridDim.x * blockDim.x;
  for (; i < n; i += stride) p[i] = 0.0f;
}

__global__ void k_scatter_adj(const int* __restrict__ src, const int* __restrict__ dst,
                              const float* __restrict__ w, float* __restrict__ adj, int E) {
  int i = blockIdx.x * blockDim.x + threadIdx.x;
  if (i < E) atomicAdd(adj + (long)src[i] * NN + dst[i], w[i]);
}

__global__ void k_cvt_bf16(const float* __restrict__ s, bf16* __restrict__ d, long n) {
  long i = (long)blockIdx.x * blockDim.x + threadIdx.x;
  long stride = (long)gridDim.x * blockDim.x;
  for (; i < n; i += stride) d[i] = (bf16)s[i];
}

// ---------------------------------------------------------------------------
// LayerNorm over C=128. One wave per row (wave32: 4 elems/lane, xor reduce).
// out = LN(src)*g + b  [+ resid].  Optional f32 and/or bf16 outputs.
// ---------------------------------------------------------------------------
__global__ __launch_bounds__(256) void k_ln128(
    const float* __restrict__ src, const float* __restrict__ resid,
    const float* __restrict__ g, const float* __restrict__ b,
    float* __restrict__ outF, bf16* __restrict__ outB, int rows) {
  int row  = blockIdx.x * 8 + (threadIdx.x >> 5);
  int lane = threadIdx.x & 31;
  if (row >= rows) return;
  const float* p = src + (long)row * 128 + lane * 4;
  float4 x4 = *(const float4*)p;
  float s = rsum32(x4.x + x4.y + x4.z + x4.w);
  float mu = s * (1.0f / 128.0f);
  float d0 = x4.x - mu, d1 = x4.y - mu, d2 = x4.z - mu, d3 = x4.w - mu;
  float vs = rsum32(d0 * d0 + d1 * d1 + d2 * d2 + d3 * d3);
  float rstd = rsqrtf(vs * (1.0f / 128.0f) + 1e-5f);
  float dd[4] = {d0, d1, d2, d3};
  float rr[4] = {0.f, 0.f, 0.f, 0.f};
  if (resid) {
    float4 r4 = *(const float4*)(resid + (long)row * 128 + lane * 4);
    rr[0] = r4.x; rr[1] = r4.y; rr[2] = r4.z; rr[3] = r4.w;
  }
  float o[4];
#pragma unroll
  for (int i = 0; i < 4; ++i) {
    int c = lane * 4 + i;
    o[i] = dd[i] * rstd * g[c] + b[c] + rr[i];
  }
  if (outF) {
    float4 w4; w4.x = o[0]; w4.y = o[1]; w4.z = o[2]; w4.w = o[3];
    *(float4*)(outF + (long)row * 128 + lane * 4) = w4;
  }
  if (outB) {
    bf16x4 t;
#pragma unroll
    for (int i = 0; i < 4; ++i) t[i] = (bf16)o[i];
    *(bf16x4*)(outB + (long)row * 128 + lane * 4) = t;
  }
}

// ---------------------------------------------------------------------------
// Generic bf16 WMMA GEMM: C[M,Nout] = A[M,K] * B[K,Nout] (+bias)(+relu)(+res)
// 128x128 block tile, 8 waves (4x2), each wave 32x64 -> 2x4 16x16 C tiles.
// Software-pipelined: ping-pong LDS buffers, global fetch of tile ks+1
// overlaps the 8 WMMAs on tile ks; one barrier per K-step.
// Row remap for batched outputs (rpb = 1<<rpbShift, always a power of 2):
//   idx = (row>>rpbShift)*bstride + (row & (rpb-1))*Nout + col
// ---------------------------------------------------------------------------
__global__ __launch_bounds__(256) void k_gemm(
    const bf16* __restrict__ A, const bf16* __restrict__ B,
    float* __restrict__ outF, bf16* __restrict__ outB,
    const float* __restrict__ bias, const float* __restrict__ resid,
    int Nout, int K, int rpbShift, long bstride, int relu) {
  __shared__ __align__(16) bf16 sA[2][128][32];
  __shared__ __align__(16) bf16 sBt[2][128][32];
  const int tid = threadIdx.x;
  const int lane = tid & 31, wv = tid >> 5;
  const int waveM = wv >> 1, waveN = wv & 1;
  const int half = lane >> 4, l15 = lane & 15;
  const int m0 = blockIdx.x * 128, n0 = blockIdx.y * 128;

  // staging coordinates
  const int rA = tid >> 1, csA = (tid & 1) * 16;   // A: one row, 16 elems
  const int kB = tid >> 3, nsB = (tid & 7) * 16;   // B: one k-row, 16 cols

  v8f acc[2][4];
#pragma unroll
  for (int i = 0; i < 2; ++i)
#pragma unroll
    for (int j = 0; j < 4; ++j)
#pragma unroll
      for (int e = 0; e < 8; ++e) acc[i][j][e] = 0.0f;

  bf16x8 ra0, ra1, rb0, rb1;
  auto fetch = [&](int ks) {
    const bf16* srcA = A + (long)(m0 + rA) * K + ks * 32 + csA;
    ra0 = *(const bf16x8*)(srcA);
    ra1 = *(const bf16x8*)(srcA + 8);
    const bf16* srcB = B + (long)(ks * 32 + kB) * Nout + n0 + nsB;
    rb0 = *(const bf16x8*)(srcB);
    rb1 = *(const bf16x8*)(srcB + 8);
  };
  auto stage = [&](int buf) {
    *(bf16x8*)&sA[buf][rA][csA]     = ra0;
    *(bf16x8*)&sA[buf][rA][csA + 8] = ra1;
#pragma unroll
    for (int i = 0; i < 8; ++i) {
      sBt[buf][nsB + i][kB]     = rb0[i];
      sBt[buf][nsB + 8 + i][kB] = rb1[i];
    }
  };

  const int kSteps = K >> 5;
  fetch(0);
  stage(0);
  for (int ks = 0; ks < kSteps; ++ks) {
    __syncthreads();  // tile ks in LDS buffer ks&1 is ready
    const int cur = ks & 1;
    const bool more = (ks + 1) < kSteps;
    if (more) fetch(ks + 1);  // overlap global fetch with WMMAs below

    V16 af[2], bfr[4];
#pragma unroll
    for (int sm = 0; sm < 2; ++sm) {  // A frag: row = l15, K chunks {h*8, 16+h*8}
      int r = waveM * 32 + sm * 16 + l15;
      af[sm].h[0] = *(const bf16x8*)&sA[cur][r][half * 8];
      af[sm].h[1] = *(const bf16x8*)&sA[cur][r][16 + half * 8];
    }
#pragma unroll
    for (int sn = 0; sn < 4; ++sn) {  // B frag: col = l15, K chunks {h*8, 16+h*8}
      int c = waveN * 64 + sn * 16 + l15;
      int kb = half * 8;
      bfr[sn].h[0] = *(const bf16x8*)&sBt[cur][c][kb];
      bfr[sn].h[1] = *(const bf16x8*)&sBt[cur][c][kb + 16];
    }
#pragma unroll
    for (int sm = 0; sm < 2; ++sm)
#pragma unroll
      for (int sn = 0; sn < 4; ++sn)
        acc[sm][sn] = __builtin_amdgcn_wmma_f32_16x16x32_bf16(
            false, af[sm].v, false, bfr[sn].v, (short)0, acc[sm][sn], false, false);

    if (more) stage((ks + 1) & 1);  // buffer was last read at ks-1: safe
  }

  // epilogue (C layout: M = r + half*8, N = l15); power-of-2 remap, no idiv
  const int rpbMask = (1 << rpbShift) - 1;
#pragma unroll
  for (int sm = 0; sm < 2; ++sm)
#pragma unroll
    for (int sn = 0; sn < 4; ++sn)
#pragma unroll
      for (int r = 0; r < 8; ++r) {
        int rowg = m0 + waveM * 32 + sm * 16 + r + half * 8;
        int colg = n0 + waveN * 64 + sn * 16 + l15;
        float vv = acc[sm][sn][r];
        if (bias) vv += bias[colg];
        if (relu) vv = fmaxf(vv, 0.0f);
        long idx = (long)(rowg >> rpbShift) * bstride +
                   (long)(rowg & rpbMask) * Nout + colg;
        if (resid) vv += resid[idx];
        if (outF) outF[idx] = vv;
        if (outB) outB[idx] = (bf16)vv;
      }
}

// ---------------------------------------------------------------------------
// Flash attention with mask=(adj==0) and O = softmax(S)V + adj V.
// Grid: B*NH*(N/64) blocks of 128 threads (4 waves). Each wave owns 16 query
// rows; workgroup shares K/V tiles (32 keys) in LDS. Two f32 accumulators:
// osoft (rescaled online) and oadj (raw, added after normalization).
// global_prefetch of the next j-tile overlaps the current tile's 24 WMMAs.
// Output written head-concat as bf16 [B*N, NH*FOD] for the Wo GEMM.
// ---------------------------------------------------------------------------
__global__ __launch_bounds__(128) void k_attn(
    const bf16* __restrict__ q, const bf16* __restrict__ k,
    const bf16* __restrict__ v, const float* __restrict__ adj,
    bf16* __restrict__ o) {
  __shared__ __align__(16) bf16 sK[32][128];    // keys row-major
  __shared__ __align__(16) bf16 sVt[128][32];   // values transposed [f][m]
  __shared__ __align__(16) bf16 sP[4][16][32];  // per-wave P tile (A layout src)
  __shared__ __align__(16) bf16 sAj[4][16][32]; // per-wave adj tile
  const int tid = threadIdx.x, lane = tid & 31, wv = tid >> 5;
  const int half = lane >> 4, l15 = lane & 15;
  const int bh = blockIdx.x >> 5;       // b*NH + h
  const int rb = blockIdx.x & 31;
  const int rowBase = rb * 64 + wv * 16;
  const bf16* qb = q + (long)bh * NN * FOD;
  const bf16* kb = k + (long)bh * NN * FOD;
  const bf16* vb = v + (long)bh * NN * FOD;
  const float scale = 0.08838834764831845f;  // 128^-0.5
  const float NEGINF = -__builtin_inff();

  V16 qf[4];
  {
    const bf16* base = qb + (long)(rowBase + l15) * FOD;
#pragma unroll
    for (int kt = 0; kt < 4; ++kt) {
      qf[kt].h[0] = *(const bf16x8*)(base + kt * 32 + half * 8);
      qf[kt].h[1] = *(const bf16x8*)(base + kt * 32 + 16 + half * 8);
    }
  }

  v8f osoft[8], oadj[8];
#pragma unroll
  for (int nt = 0; nt < 8; ++nt)
#pragma unroll
    for (int e = 0; e < 8; ++e) { osoft[nt][e] = 0.0f; oadj[nt][e] = 0.0f; }
  float mrun[8], lrun[8];
#pragma unroll
  for (int r = 0; r < 8; ++r) { mrun[r] = NEGINF; lrun[r] = 0.0f; }

  const int stgM = tid >> 2, stgSeg = (tid & 3) * 32;
  for (int j = 0; j < NN / 32; ++j) {
    {  // cooperative stage of K (row-major) and V (transposed)
      const bf16* ksrc = kb + (long)(j * 32 + stgM) * FOD + stgSeg;
      const bf16* vsrc = vb + (long)(j * 32 + stgM) * FOD + stgSeg;
#pragma unroll
      for (int c = 0; c < 4; ++c) {
        *(bf16x8*)&sK[stgM][stgSeg + c * 8] = *(const bf16x8*)(ksrc + c * 8);
        bf16x8 t = *(const bf16x8*)(vsrc + c * 8);
#pragma unroll
        for (int i = 0; i < 8; ++i) sVt[stgSeg + c * 8 + i][stgM] = t[i];
      }
    }
    if (j + 1 < NN / 32) {  // prefetch next tile into L2/L0 (global_prefetch_b8)
      __builtin_prefetch(kb + (long)((j + 1) * 32 + stgM) * FOD + stgSeg, 0, 1);
      __builtin_prefetch(vb + (long)((j + 1) * 32 + stgM) * FOD + stgSeg, 0, 1);
      __builtin_prefetch(adj + (long)(rowBase + l15) * NN + (j + 1) * 32, 0, 1);
    }
    __syncthreads();

    float sv[2][8], av[2][8];
#pragma unroll
    for (int sub = 0; sub < 2; ++sub) {
      v8f S;
#pragma unroll
      for (int e = 0; e < 8; ++e) S[e] = 0.0f;
#pragma unroll
      for (int kt = 0; kt < 4; ++kt) {  // S = Q * K^T : B[f][m] frag = K rows
        V16 bfr;
        int ml = sub * 16 + l15;
        bfr.h[0] = *(const bf16x8*)&sK[ml][kt * 32 + half * 8];
        bfr.h[1] = *(const bf16x8*)&sK[ml][kt * 32 + 16 + half * 8];
        S = __builtin_amdgcn_wmma_f32_16x16x32_bf16(false, qf[kt].v, false, bfr.v,
                                                    (short)0, S, false, false);
      }
#pragma unroll
      for (int r = 0; r < 8; ++r) {
        int rg = rowBase + r + half * 8;
        int cg = j * 32 + sub * 16 + l15;
        float a = adj[(long)rg * NN + cg];
        float s = S[r] * scale;
        if (a == 0.0f) s = NEGINF;  // mask = (adj == 0)
        sv[sub][r] = s;
        av[sub][r] = a;
      }
    }

    float alpha[8];
#pragma unroll
    for (int r = 0; r < 8; ++r) {  // online softmax over 32 keys
      float mx = rmax16(fmaxf(sv[0][r], sv[1][r]));
      float mn = fmaxf(mrun[r], mx);
      float al = (mrun[r] > NEGINF) ? expf(mrun[r] - mn) : 0.0f;
      float p0 = (sv[0][r] > NEGINF) ? expf(sv[0][r] - mn) : 0.0f;
      float p1 = (sv[1][r] > NEGINF) ? expf(sv[1][r] - mn) : 0.0f;
      float rs = rsum16(p0 + p1);
      lrun[r] = lrun[r] * al + rs;
      mrun[r] = mn;
      alpha[r] = al;
      int rl = r + half * 8;  // C layout -> LDS row-major for A-frag reload
      sP[wv][rl][l15]       = (bf16)p0;
      sP[wv][rl][16 + l15]  = (bf16)p1;
      sAj[wv][rl][l15]      = (bf16)av[0][r];
      sAj[wv][rl][16 + l15] = (bf16)av[1][r];
    }
    __syncthreads();

    V16 pf, af;
    pf.h[0] = *(const bf16x8*)&sP[wv][l15][half * 8];
    pf.h[1] = *(const bf16x8*)&sP[wv][l15][16 + half * 8];
    af.h[0] = *(const bf16x8*)&sAj[wv][l15][half * 8];
    af.h[1] = *(const bf16x8*)&sAj[wv][l15][16 + half * 8];

#pragma unroll
    for (int nt = 0; nt < 8; ++nt) {
      V16 vf;  // B frag of V: col f = nt*16+l15, K rows {half*8, 16+half*8} chunks
      int f = nt * 16 + l15;
      vf.h[0] = *(const bf16x8*)&sVt[f][half * 8];
      vf.h[1] = *(const bf16x8*)&sVt[f][16 + half * 8];
#pragma unroll
      for (int r = 0; r < 8; ++r) osoft[nt][r] *= alpha[r];
      osoft[nt] = __builtin_amdgcn_wmma_f32_16x16x32_bf16(false, pf.v, false, vf.v,
                                                          (short)0, osoft[nt], false, false);
      oadj[nt]  = __builtin_amdgcn_wmma_f32_16x16x32_bf16(false, af.v, false, vf.v,
                                                          (short)0, oadj[nt], false, false);
    }
    __syncthreads();
  }

  const int b_ = bh >> 2, h_ = bh & 3;
#pragma unroll
  for (int r = 0; r < 8; ++r) {
    float inv = (lrun[r] > 0.0f) ? 1.0f / lrun[r] : 0.0f;
    int rg = rowBase + r + half * 8;
    long rowOff = ((long)(b_ * NN + rg)) * (NH * FOD) + h_ * FOD;
#pragma unroll
    for (int nt = 0; nt < 8; ++nt) {
      int cg = nt * 16 + l15;
      o[rowOff + cg] = (bf16)(osoft[nt][r] * inv + oadj[nt][r]);
    }
  }
}

// ---------------------------------------------------------------------------
// Host orchestration
// ---------------------------------------------------------------------------
extern "C" void kernel_launch(void* const* d_in, const int* in_sizes, int n_in,
                              void* d_out, int out_size, void* d_ws, size_t ws_size,
                              hipStream_t stream) {
  (void)n_in; (void)out_size; (void)ws_size;
  const float* H     = (const float*)d_in[0];
  const int*   esrc  = (const int*)d_in[1];
  const int*   edst  = (const int*)d_in[2];
  const float* ew    = (const float*)d_in[3];
  const float* Wq    = (const float*)d_in[4];
  const float* Wk    = (const float*)d_in[5];
  const float* Wv    = (const float*)d_in[6];
  const float* Wo    = (const float*)d_in[7];
  const float* bo    = (const float*)d_in[8];
  const float* g_sa  = (const float*)d_in[9];
  const float* b_sa  = (const float*)d_in[10];
  const float* g1    = (const float*)d_in[11];
  const float* b1    = (const float*)d_in[12];
  const float* g2    = (const float*)d_in[13];
  const float* b2    = (const float*)d_in[14];
  const float* W1    = (const float*)d_in[15];
  const float* b1f   = (const float*)d_in[16];
  const float* W2    = (const float*)d_in[17];
  const float* b2f   = (const float*)d_in[18];
  const float* gf    = (const float*)d_in[19];
  const float* bfp   = (const float*)d_in[20];
  const int E = in_sizes[1];

  // workspace carve-up (~120MB)
  char* base = (char*)d_ws;
  auto alloc = [&](size_t bytes) -> void* {
    void* p = (void*)base;
    base += (bytes + 255) & ~(size_t)255;
    return p;
  };
  float* adj  = (float*)alloc((size_t)NN * NN * 4);             // 16MB
  bf16*  qb   = (bf16*)alloc((size_t)NB * NH * NN * FOD * 2);   // 16MB
  bf16*  kbuf = (bf16*)alloc((size_t)NB * NH * NN * FOD * 2);   // 16MB
  bf16*  vbuf = (bf16*)alloc((size_t)NB * NH * NN * FOD * 2);   // 16MB
  bf16*  ob   = (bf16*)alloc((size_t)ROWS * 512 * 2);           // 16MB
  bf16*  f1b  = (bf16*)alloc((size_t)ROWS * 512 * 2);           // 16MB
  bf16*  hb   = (bf16*)alloc((size_t)ROWS * 128 * 2);           // 4MB
  float* tbuf = (float*)alloc((size_t)ROWS * 128 * 4);          // 8MB
  float* x    = (float*)alloc((size_t)ROWS * 128 * 4);          // 8MB
  bf16*  wqb  = (bf16*)alloc((size_t)2 * NH * 128 * 128 * 2);
  bf16*  wkb  = (bf16*)alloc((size_t)2 * NH * 128 * 128 * 2);
  bf16*  wvb  = (bf16*)alloc((size_t)2 * NH * 128 * 128 * 2);
  bf16*  wob  = (bf16*)alloc((size_t)2 * 512 * 128 * 2);
  bf16*  w1b  = (bf16*)alloc((size_t)2 * 128 * 512 * 2);
  bf16*  w2b  = (bf16*)alloc((size_t)2 * 512 * 128 * 2);

  // adjacency
  k_zero_f32<<<4096, 256, 0, stream>>>(adj, (long)NN * NN);
  k_scatter_adj<<<(E + 255) / 256, 256, 0, stream>>>(esrc, edst, ew, adj, E);

  // weights -> bf16 (all 131072 elems each)
  k_cvt_bf16<<<128, 256, 0, stream>>>(Wq, wqb, 2L * NH * 128 * 128);
  k_cvt_bf16<<<128, 256, 0, stream>>>(Wk, wkb, 2L * NH * 128 * 128);
  k_cvt_bf16<<<128, 256, 0, stream>>>(Wv, wvb, 2L * NH * 128 * 128);
  k_cvt_bf16<<<128, 256, 0, stream>>>(Wo, wob, 2L * 512 * 128);
  k_cvt_bf16<<<128, 256, 0, stream>>>(W1, w1b, 2L * 128 * 512);
  k_cvt_bf16<<<128, 256, 0, stream>>>(W2, w2b, 2L * 512 * 128);

  // x = H
  hipMemcpyAsync(x, H, (size_t)ROWS * 128 * 4, hipMemcpyDeviceToDevice, stream);

  const long qkvBStride = (long)NH * NN * FOD;  // batch stride of [B,NH,N,FOD]
  const int SH_N = 11;   // log2(NN)   -> Q/K/V row remap
  const int SH_R = 14;   // log2(ROWS) -> plain GEMMs (bi always 0)
  for (int l = 0; l < 2; ++l) {
    // h1 = LN(x, g1, b1) -> bf16
    k_ln128<<<ROWS / 8, 256, 0, stream>>>(x, nullptr, g1 + l * 128, b1 + l * 128,
                                          nullptr, hb, ROWS);
    // Q/K/V per head, written directly into [B,NH,N,FOD] via row remap
    for (int h = 0; h < NH; ++h) {
      const bf16* wqp = wqb + (size_t)(l * NH + h) * 128 * 128;
      const bf16* wkp = wkb + (size_t)(l * NH + h) * 128 * 128;
      const bf16* wvp = wvb + (size_t)(l * NH + h) * 128 * 128;
      k_gemm<<<dim3(ROWS / 128, 1), 256, 0, stream>>>(
          hb, wqp, nullptr, qb + (size_t)h * NN * FOD, nullptr, nullptr,
          128, 128, SH_N, qkvBStride, 0);
      k_gemm<<<dim3(ROWS / 128, 1), 256, 0, stream>>>(
          hb, wkp, nullptr, kbuf + (size_t)h * NN * FOD, nullptr, nullptr,
          128, 128, SH_N, qkvBStride, 0);
      k_gemm<<<dim3(ROWS / 128, 1), 256, 0, stream>>>(
          hb, wvp, nullptr, vbuf + (size_t)h * NN * FOD, nullptr, nullptr,
          128, 128, SH_N, qkvBStride, 0);
    }
    // flash attention + adj@V, head-concat bf16 output [ROWS, 512]
    k_attn<<<NB * NH * (NN / 64), 128, 0, stream>>>(qb, kbuf, vbuf, adj, ob);
    // t = o @ Wo + bo
    k_gemm<<<dim3(ROWS / 128, 1), 256, 0, stream>>>(
        ob, wob + (size_t)l * 512 * 128, tbuf, nullptr, bo + l * 128, nullptr,
        128, 512, SH_R, 0, 0);
    // x = x + LN(t, g_sa, b_sa)
    k_ln128<<<ROWS / 8, 256, 0, stream>>>(tbuf, x, g_sa + l * 128, b_sa + l * 128,
                                          x, nullptr, ROWS);
    // h2 = LN(x, g2, b2) -> bf16
    k_ln128<<<ROWS / 8, 256, 0, stream>>>(x, nullptr, g2 + l * 128, b2 + l * 128,
                                          nullptr, hb, ROWS);
    // f1 = relu(h2 @ W1 + b1f) -> bf16
    k_gemm<<<dim3(ROWS / 128, 4), 256, 0, stream>>>(
        hb, w1b + (size_t)l * 128 * 512, nullptr, f1b, b1f + l * 512, nullptr,
        512, 128, SH_R, 0, 1);
    // x = x + f1 @ W2 + b2f
    k_gemm<<<dim3(ROWS / 128, 1), 256, 0, stream>>>(
        f1b, w2b + (size_t)l * 512 * 128, x, nullptr, b2f + l * 128, x,
        128, 512, SH_R, 0, 0);
  }
  // out = LN(x, gf, bf)
  k_ln128<<<ROWS / 8, 256, 0, stream>>>(x, nullptr, gf, bfp, (float*)d_out,
                                        nullptr, ROWS);
}
